// StereoMerging_60567628808608
// MI455X (gfx1250) — compile-verified, matrix operand order checked by reading, not run on hardware
//
#include <hip/hip_runtime.h>
#include <hip/hip_bf16.h>

typedef __attribute__((ext_vector_type(16))) _Float16 v16h;
typedef __attribute__((ext_vector_type(8)))  float    v8f;

#define CONV_RELU 1
#define CONV_UP2  2

__device__ __forceinline__ int imin(int a, int b) { return a < b ? a : b; }
__device__ __forceinline__ int imax(int a, int b) { return a > b ? a : b; }

// ---------------------------------------------------------------------------
// Implicit-GEMM conv2d on the WMMA path (f16 in, f32 accumulate).
//   GEMM: M = Cout, N = B*Hout*Wout, K = Cin*KH*KW
//   Block = 256 threads (8 waves), tile BM=64 x BN=128, K step 32.
//   Wave (m,n) grid: 4 M-strips x 2 N-halves; 4 WMMA tiles per wave.
//   Double-buffered LDS; vectorized float4 weight staging; pointer-array
//   im2col gather (distinct address regs -> loads stay in flight).
//   Fused bias/BN/residual/ReLU and 2x bilinear-upsample gather.
// ---------------------------------------------------------------------------
struct ConvP {
  const float* in;   // [B, inStride, Hin, Win] (source dims even when UP2)
  const float* wgt;  // [Cout, Cin*KH*KW] contiguous
  const float* bias; // [Cout] or null
  const float* bns;  // bn scale [Cout] or null
  const float* bno;  // bn offset [Cout] or null
  const float* res;  // residual [B, resStride, Hout, Wout] or null
  float*       out;  // [B, outStride, Hout, Wout]
  int B, Cin, Hin, Win, Cout, Hout, Wout, KH, KW, pad;
  int inOff, inStride, outOff, outStride, resStride;
  int flags;
};

// Bilinear (align_corners) fetch for the fused-upsample path, branch-free.
__device__ __forceinline__ float up2_fetch(const ConvP& p, int b, int ci,
                                           int hi, int wi, int Hup, int Wup,
                                           float sy, float sx) {
  const bool inb = ((unsigned)hi < (unsigned)Hup) && ((unsigned)wi < (unsigned)Wup);
  const int hc = imin(imax(hi, 0), Hup - 1);
  const int wc = imin(imax(wi, 0), Wup - 1);
  float fy = (float)hc * sy, fx = (float)wc * sx;
  int y0 = (int)fy, x0 = (int)fx;
  int y1 = imin(y0 + 1, p.Hin - 1), x1 = imin(x0 + 1, p.Win - 1);
  float wy = fy - (float)y0, wx = fx - (float)x0;
  const float* base = p.in + ((size_t)b * p.inStride + p.inOff + ci) * (size_t)(p.Hin * p.Win);
  float v00 = base[y0 * p.Win + x0], v01 = base[y0 * p.Win + x1];
  float v10 = base[y1 * p.Win + x0], v11 = base[y1 * p.Win + x1];
  float v = (v00 * (1.f - wx) + v01 * wx) * (1.f - wy) +
            (v10 * (1.f - wx) + v11 * wx) * wy;
  return inb ? v : 0.f;
}

// Stage one K-tile: A 64x32 (weights), B 128x32 (im2col), into LDS.
__device__ __forceinline__ void stage_tile(
    const ConvP& p, int k0, _Float16* sa, _Float16* sb,
    int rowA, int kbA, int rowB, int kbB,
    const float* wrowA, bool MAok, int K, int KHW,
    bool nBok, int bb, int ho, int wo, int Hup, int Wup, float sy, float sx) {
  // ---- A: weights, 8 consecutive k per thread (vectorized fast path) ----
  {
    union H8 { _Float16 h[8]; uint4 u4; } pk;
    int kk0 = k0 + kbA;
    if (MAok && kk0 + 8 <= K) {
      const float4* p4 = (const float4*)(wrowA + kk0);   // 16B aligned: K%4==0
      float4 a0 = p4[0], a1 = p4[1];
      pk.h[0] = (_Float16)a0.x; pk.h[1] = (_Float16)a0.y;
      pk.h[2] = (_Float16)a0.z; pk.h[3] = (_Float16)a0.w;
      pk.h[4] = (_Float16)a1.x; pk.h[5] = (_Float16)a1.y;
      pk.h[6] = (_Float16)a1.z; pk.h[7] = (_Float16)a1.w;
    } else {
#pragma unroll
      for (int i = 0; i < 8; ++i) {
        int kk = kk0 + i;
        float v = wrowA[kk < K ? kk : 0];
        pk.h[i] = (_Float16)((MAok && kk < K) ? v : 0.f);
      }
    }
    *(uint4*)&sa[rowA * 32 + kbA] = pk.u4;               // ds_store_b128
  }
  // ---- B: im2col gather, 16 consecutive k per thread ----
  union H16 { _Float16 h[16]; uint4 u4[2]; } pb;
  if (p.flags & CONV_UP2) {
#pragma unroll
    for (int i = 0; i < 16; ++i) {
      int kk = k0 + kbB + i;
      int kc = kk < K ? kk : 0;
      int ci = kc / KHW;
      int rr = kc - ci * KHW;
      int kh = rr / p.KW;
      int kw = rr - kh * p.KW;
      float v = up2_fetch(p, bb, ci, ho + kh - p.pad, wo + kw - p.pad, Hup, Wup, sy, sx);
      pb.h[i] = (_Float16)((nBok && kk < K) ? v : 0.f);
    }
  } else {
    // phase 1: addresses into distinct registers (no xcnt serialization)
    const float* aptr[16];
    unsigned msk = 0;
#pragma unroll
    for (int i = 0; i < 16; ++i) {
      int kk = k0 + kbB + i;
      int kc = kk < K ? kk : 0;
      int ci = kc / KHW;
      int rr = kc - ci * KHW;
      int kh = rr / p.KW;
      int kw = rr - kh * p.KW;
      int hi = ho + kh - p.pad, wi = wo + kw - p.pad;
      bool inb = nBok && (kk < K) &&
                 ((unsigned)hi < (unsigned)p.Hin) && ((unsigned)wi < (unsigned)p.Win);
      int hc = imin(imax(hi, 0), p.Hin - 1);
      int wc = imin(imax(wi, 0), p.Win - 1);
      aptr[i] = p.in + (((size_t)bb * p.inStride + p.inOff + ci) * p.Hin + hc) * p.Win + wc;
      msk |= (inb ? 1u : 0u) << i;
    }
    // phase 2: unconditional loads, batched in flight
    float vb[16];
#pragma unroll
    for (int i = 0; i < 16; ++i) vb[i] = *aptr[i];
#pragma unroll
    for (int i = 0; i < 16; ++i)
      pb.h[i] = (_Float16)(((msk >> i) & 1u) ? vb[i] : 0.f);
  }
  uint4* dst = (uint4*)&sb[rowB * 32 + kbB];
  dst[0] = pb.u4[0];
  dst[1] = pb.u4[1];
}

__global__ __launch_bounds__(256) void k_conv2d_wmma(ConvP p) {
  __shared__ _Float16 sA[2][64 * 32];    // [m_local][k_local]  weights
  __shared__ _Float16 sB[2][128 * 32];   // [n_local][k_local]  im2col

  const int tid   = threadIdx.x;
  const int wave  = tid >> 5;
  const int lane  = tid & 31;
  const int half  = lane >> 4;
  const int lrow  = lane & 15;
  const int waveM = wave & 3;            // 4 M-strips of 16
  const int waveN = wave >> 2;           // 2 N-halves of 64

  const int KHW = p.KH * p.KW;
  const int K   = p.Cin * KHW;
  const int N   = p.B * p.Hout * p.Wout;
  const int m0  = blockIdx.y * 64;
  const int n0  = blockIdx.x * 128;

  // staging assignment
  const int rowA = tid >> 2;             // 0..63,  4 threads/row, 8 k each
  const int kbA  = (tid & 3) * 8;
  const int rowB = tid >> 1;             // 0..127, 2 threads/row, 16 k each
  const int kbB  = (tid & 1) * 16;

  // output-pixel coordinates for this thread's B row (clamped if OOB)
  const int nB = n0 + rowB;
  const bool nBok = (nB < N);
  const int nC = nBok ? nB : 0;
  const int wo = nC % p.Wout;
  const int ho = (nC / p.Wout) % p.Hout;
  const int bb = nC / (p.Wout * p.Hout);

  const int Hup = (p.flags & CONV_UP2) ? 2 * p.Hin : p.Hin;
  const int Wup = (p.flags & CONV_UP2) ? 2 * p.Win : p.Win;
  const float sy = (Hup > 1) ? (float)(p.Hin - 1) / (float)(Hup - 1) : 0.f;
  const float sx = (Wup > 1) ? (float)(p.Win - 1) / (float)(Wup - 1) : 0.f;

  const int  MA   = m0 + rowA;
  const bool MAok = (MA < p.Cout);
  const float* wrowA = p.wgt + (size_t)(MAok ? MA : 0) * K;

  v8f acc[4];
#pragma unroll
  for (int t = 0; t < 4; ++t)
#pragma unroll
    for (int i = 0; i < 8; ++i) acc[t][i] = 0.f;

  const int KT = (K + 31) / 32;

  // prologue: stage K-tile 0 into buffer 0
  stage_tile(p, 0, sA[0], sB[0], rowA, kbA, rowB, kbB,
             wrowA, MAok, K, KHW, nBok, bb, ho, wo, Hup, Wup, sy, sx);
  __syncthreads();

  for (int kt = 0; kt < KT; ++kt) {
    const int cur = kt & 1;

    // overlap: stage next K-tile into the other buffer while WMMAs issue
    if (kt + 1 < KT) {
      if (kt + 2 < KT)
        __builtin_prefetch(wrowA + (kt + 2) * 32, 0, 1);   // global_prefetch_b8
      stage_tile(p, (kt + 1) * 32, sA[cur ^ 1], sB[cur ^ 1], rowA, kbA, rowB, kbB,
                 wrowA, MAok, K, KHW, nBok, bb, ho, wo, Hup, Wup, sy, sx);
    }

    // ---- A fragment per documented CDNA5 16-bit 16x32 layout ----
    union Frag { v16h h; unsigned u[8]; };
    Frag af;
    const unsigned* pa = (const unsigned*)&sA[cur][(waveM * 16 + lrow) * 32];
#pragma unroll
    for (int v = 0; v < 8; ++v) {
      int kk = ((v >> 2) << 4) + half * 8 + ((v & 3) << 1);
      af.u[v] = pa[kk >> 1];
    }
    // ---- 4 N-subtiles in this wave's half: B fragment + WMMA ----
#pragma unroll
    for (int t = 0; t < 4; ++t) {
      Frag bf;
      const unsigned* pb = (const unsigned*)&sB[cur][(waveN * 64 + t * 16 + lrow) * 32];
#pragma unroll
      for (int v = 0; v < 8; ++v) {
        int kk = ((v >> 2) << 4) + half * 8 + ((v & 3) << 1);
        bf.u[v] = pb[kk >> 1];
      }
      acc[t] = __builtin_amdgcn_wmma_f32_16x16x32_f16(
          false, af.h, false, bf.h, (short)0, acc[t], false, false);
    }
    __syncthreads();   // staging of cur^1 done; reads of cur done
  }

  // ---- fused epilogue: bias + BN affine + residual + ReLU + offset store ----
#pragma unroll
  for (int t = 0; t < 4; ++t) {
#pragma unroll
    for (int r = 0; r < 8; ++r) {
      int co = m0 + waveM * 16 + r + half * 8;
      int n  = n0 + waveN * 64 + t * 16 + lrow;
      if (co < p.Cout && n < N) {
        int w = n % p.Wout; int rr = n / p.Wout;
        int h = rr % p.Hout; int b = rr / p.Hout;
        float v = acc[t][r];
        if (p.bias) v += p.bias[co];
        if (p.bns)  v = v * p.bns[co] + p.bno[co];
        if (p.res)  v += p.res[(((size_t)b * p.resStride + co) * p.Hout + h) * p.Wout + w];
        if (p.flags & CONV_RELU) v = v > 0.f ? v : 0.f;
        p.out[(((size_t)b * p.outStride + p.outOff + co) * p.Hout + h) * p.Wout + w] = v;
      }
    }
  }
}

// ---------------------------------------------------------------------------
// PSM cosine cost volume: out[b,d,h,w] = mean_c L[b,c,h,w]*R[b,c,h,w-d], w>=d
// ---------------------------------------------------------------------------
__global__ void k_psm(const float* __restrict__ L, const float* __restrict__ R,
                      float* __restrict__ out, int B, int C, int H, int W, int D) {
  int i = blockIdx.x * blockDim.x + threadIdx.x;
  int total = B * D * H * W;
  if (i >= total) return;
  int w = i % W; int h = (i / W) % H; int d = (i / (W * H)) % D; int b = i / (W * H * D);
  float acc = 0.f;
  if (w >= d) {
    const float* lp = L + ((size_t)b * C * H + h) * W + w;
    const float* rp = R + ((size_t)b * C * H + h) * W + w - d;
    for (int c = 0; c < C; ++c) acc += lp[(size_t)c * H * W] * rp[(size_t)c * H * W];
    acc *= (1.f / (float)C);
  }
  out[i] = acc;
}

// ---------------------------------------------------------------------------
// Shifted-concat 3D cost volume: [B,16,D,H,W] from lf/rf [B,8,H,W]
// ---------------------------------------------------------------------------
__global__ void k_cost3d(const float* __restrict__ lf, const float* __restrict__ rf,
                         float* __restrict__ out, int B, int H, int W, int D) {
  int i = blockIdx.x * blockDim.x + threadIdx.x;
  int total = B * 16 * D * H * W;
  if (i >= total) return;
  int w = i % W; int h = (i / W) % H; int d = (i / (W * H)) % D;
  int ch = (i / (W * H * D)) % 16; int b = i / (W * H * D * 16);
  float v = 0.f;
  if (w >= d) {
    if (ch < 8) v = lf[(((size_t)b * 8 + ch) * H + h) * W + w];
    else        v = rf[(((size_t)b * 8 + (ch - 8)) * H + h) * W + (w - d)];
  }
  out[i] = v;
}

// ---------------------------------------------------------------------------
// Direct 3x3x3 conv3d + bias + BN + ReLU (tiny channel counts)
// ---------------------------------------------------------------------------
__global__ void k_conv3d(const float* __restrict__ in, const float* __restrict__ wgt,
                         const float* __restrict__ bias, const float* __restrict__ s,
                         const float* __restrict__ o, float* __restrict__ out,
                         int B, int Ci, int Co, int D, int H, int W) {
  int i = blockIdx.x * blockDim.x + threadIdx.x;
  int total = B * Co * D * H * W;
  if (i >= total) return;
  int w = i % W; int h = (i / W) % H; int d = (i / (W * H)) % D;
  int co = (i / (W * H * D)) % Co; int b = i / (W * H * D * Co);
  float acc = bias[co];
  for (int ci = 0; ci < Ci; ++ci) {
    const float* ip = in + (((size_t)b * Ci + ci) * D) * H * W;
    const float* wp = wgt + (((size_t)co * Ci + ci) * 27);
    for (int kd = 0; kd < 3; ++kd) {
      int dd = d + kd - 1; if ((unsigned)dd >= (unsigned)D) continue;
      for (int kh = 0; kh < 3; ++kh) {
        int hh = h + kh - 1; if ((unsigned)hh >= (unsigned)H) continue;
        for (int kw = 0; kw < 3; ++kw) {
          int ww = w + kw - 1; if ((unsigned)ww >= (unsigned)W) continue;
          acc += ip[((size_t)dd * H + hh) * W + ww] * wp[(kd * 3 + kh) * 3 + kw];
        }
      }
    }
  }
  float v = acc * s[co] + o[co];
  out[i] = v > 0.f ? v : 0.f;
}

// ---------------------------------------------------------------------------
// Depthwise 3x3 + BN + ReLU, channel-offset in/out (ghost modules)
// ---------------------------------------------------------------------------
__global__ void k_dwconv(const float* __restrict__ in, const float* __restrict__ wgt,
                         const float* __restrict__ s, const float* __restrict__ o,
                         float* __restrict__ out, int B, int C, int H, int W,
                         int inOff, int inStride, int outOff, int outStride) {
  int i = blockIdx.x * blockDim.x + threadIdx.x;
  int total = B * C * H * W;
  if (i >= total) return;
  int w = i % W; int h = (i / W) % H; int c = (i / (W * H)) % C; int b = i / (W * H * C);
  const float* ip = in + ((size_t)b * inStride + inOff + c) * (size_t)(H * W);
  const float* wp = wgt + (size_t)c * 9;
  float acc = 0.f;
  for (int kh = 0; kh < 3; ++kh) {
    int hh = h + kh - 1; if ((unsigned)hh >= (unsigned)H) continue;
    for (int kw = 0; kw < 3; ++kw) {
      int ww = w + kw - 1; if ((unsigned)ww >= (unsigned)W) continue;
      acc += ip[hh * W + ww] * wp[kh * 3 + kw];
    }
  }
  float v = acc * s[c] + o[c];
  out[(((size_t)b * outStride + outOff + c) * H + h) * W + w] = v > 0.f ? v : 0.f;
}

// ---------------------------------------------------------------------------
// 2x2 average pool
// ---------------------------------------------------------------------------
__global__ void k_avgpool2(const float* __restrict__ in, float* __restrict__ out,
                           int B, int C, int H, int W) {
  int Ho = H >> 1, Wo = W >> 1;
  int i = blockIdx.x * blockDim.x + threadIdx.x;
  int total = B * C * Ho * Wo;
  if (i >= total) return;
  int w = i % Wo; int h = (i / Wo) % Ho; int c = (i / (Wo * Ho)) % C; int b = i / (Wo * Ho * C);
  const float* ip = in + (((size_t)b * C + c) * H + 2 * h) * W + 2 * w;
  out[i] = 0.25f * (ip[0] + ip[1] + ip[W] + ip[W + 1]);
}

// ---------------------------------------------------------------------------
// Channel-offset copy (concats)
// ---------------------------------------------------------------------------
__global__ void k_copy_chan(const float* __restrict__ src, float* __restrict__ dst,
                            int B, int C, int H, int W, int off, int stride) {
  int i = blockIdx.x * blockDim.x + threadIdx.x;
  int total = B * C * H * W;
  if (i >= total) return;
  int w = i % W; int h = (i / W) % H; int c = (i / (W * H)) % C; int b = i / (W * H * C);
  dst[(((size_t)b * stride + off + c) * H + h) * W + w] = src[i];
}

// ---------------------------------------------------------------------------
// Orchestration
// ---------------------------------------------------------------------------
extern "C" void kernel_launch(void* const* d_in, const int* in_sizes, int n_in,
                              void* d_out, int out_size, void* d_ws, size_t ws_size,
                              hipStream_t stream) {
  (void)in_sizes; (void)n_in; (void)out_size; (void)ws_size;
  const int B = 4;
  const float* left0  = (const float*)d_in[0];
  const float* left1  = (const float*)d_in[1];
  const float* left2  = (const float*)d_in[2];
  const float* right0 = (const float*)d_in[3];
  const float* right1 = (const float*)d_in[4];
  const float* right2 = (const float*)d_in[5];
  // params flattened in setup_inputs insertion order, starting at d_in[6]
  const float* P[64];
  for (int i = 6; i < 64; ++i) P[i] = (const float*)d_in[i];

  float* ws = (float*)d_ws;
  size_t off = 0;
  auto alloc = [&](size_t n) { float* p = ws + off; off += n; return p; };
  float* psv0   = alloc((size_t)B * 12 * 96 * 320);
  float* psv1   = alloc((size_t)B * 12 * 48 * 160);
  float* lf     = alloc((size_t)B * 8 * 24 * 80);
  float* rf     = alloc((size_t)B * 8 * 24 * 80);
  float* cost3d = alloc((size_t)B * 16 * 6 * 24 * 80);
  float* c3a    = alloc((size_t)B * 8 * 6 * 24 * 80);
  float* c3b    = alloc((size_t)B * 8 * 6 * 24 * 80);   // == psv2 [B,48,24,80]
  float* g0cat  = alloc((size_t)B * 36 * 96 * 320);
  float* pool0  = alloc((size_t)B * 36 * 48 * 160);
  float* b0tmp  = alloc((size_t)B * 36 * 48 * 160);
  float* cat48  = alloc((size_t)B * 48 * 48 * 160);
  float* g1cat  = alloc((size_t)B * 144 * 48 * 160);
  float* pool1  = alloc((size_t)B * 144 * 24 * 80);
  float* b1tmp  = alloc((size_t)B * 144 * 24 * 80);
  float* cat192 = alloc((size_t)B * 192 * 24 * 80);
  float* g2cat  = alloc((size_t)B * 576 * 24 * 80);
  float* b2tmp  = alloc((size_t)B * 576 * 24 * 80);
  float* d1out  = alloc((size_t)B * 288 * 48 * 160);
  float* d2out  = alloc((size_t)B * 144 * 96 * 320);

  float* outF = (float*)d_out;                         // features [B,704,24,80]
  float* outD = outF + (size_t)B * 704 * 24 * 80;      // depth    [B,48,96,320]

  auto conv = [&](ConvP p) {
    int N = p.B * p.Hout * p.Wout;
    dim3 g((N + 127) / 128, (p.Cout + 63) / 64);
    k_conv2d_wmma<<<g, dim3(256), 0, stream>>>(p);
  };
  auto g1d = [](int n) { return dim3((n + 255) / 256); };

  // --- PSM cost volumes (levels 0, 1) ---
  k_psm<<<g1d(B * 12 * 96 * 320), 256, 0, stream>>>(left0, right0, psv0, B, 32, 96, 320, 12);
  k_psm<<<g1d(B * 12 * 48 * 160), 256, 0, stream>>>(left1, right1, psv1, B, 64, 48, 160, 12);

  // --- CostVolume @1/16: 1x1 downsample 128->8 (WMMA), shifted concat, 2x conv3d ---
  conv({left2, P[6], P[7], P[8], P[9], nullptr, lf,
        B, 128, 24, 80, 8, 24, 80, 1, 1, 0, 0, 128, 0, 8, 0, CONV_RELU});
  conv({right2, P[6], P[7], P[8], P[9], nullptr, rf,
        B, 128, 24, 80, 8, 24, 80, 1, 1, 0, 0, 128, 0, 8, 0, CONV_RELU});
  k_cost3d<<<g1d(B * 16 * 6 * 24 * 80), 256, 0, stream>>>(lf, rf, cost3d, B, 24, 80, 6);
  k_conv3d<<<g1d(B * 8 * 6 * 24 * 80), 256, 0, stream>>>(cost3d, P[10], P[11], P[12], P[13], c3a, B, 16, 8, 6, 24, 80);
  k_conv3d<<<g1d(B * 8 * 6 * 24 * 80), 256, 0, stream>>>(c3a, P[14], P[15], P[16], P[17], c3b, B, 8, 8, 6, 24, 80);

  // --- ghost0 (12 -> 36) ---
  k_copy_chan<<<g1d(B * 12 * 96 * 320), 256, 0, stream>>>(psv0, g0cat, B, 12, 96, 320, 0, 36);
  conv({psv0, P[18], nullptr, P[19], P[20], nullptr, g0cat,
        B, 12, 96, 320, 12, 96, 320, 3, 3, 1, 0, 12, 12, 36, 0, CONV_RELU});
  k_dwconv<<<g1d(B * 12 * 96 * 320), 256, 0, stream>>>(g0cat, P[21], P[22], P[23], g0cat, B, 12, 96, 320, 12, 36, 24, 36);
  k_avgpool2<<<g1d(B * 36 * 48 * 160), 256, 0, stream>>>(g0cat, pool0, B, 36, 96, 320);

  // --- basic b0 (36) -> cat with psv1 (48) ---
  conv({pool0, P[36], nullptr, P[37], P[38], nullptr, b0tmp,
        B, 36, 48, 160, 36, 48, 160, 3, 3, 1, 0, 36, 0, 36, 0, CONV_RELU});
  conv({b0tmp, P[39], nullptr, P[40], P[41], pool0, cat48,
        B, 36, 48, 160, 36, 48, 160, 3, 3, 1, 0, 36, 0, 48, 36, CONV_RELU});
  k_copy_chan<<<g1d(B * 12 * 48 * 160), 256, 0, stream>>>(psv1, cat48, B, 12, 48, 160, 36, 48);

  // --- ghost1 (48 -> 144) ---
  k_copy_chan<<<g1d(B * 48 * 48 * 160), 256, 0, stream>>>(cat48, g1cat, B, 48, 48, 160, 0, 144);
  conv({cat48, P[24], nullptr, P[25], P[26], nullptr, g1cat,
        B, 48, 48, 160, 48, 48, 160, 3, 3, 1, 0, 48, 48, 144, 0, CONV_RELU});
  k_dwconv<<<g1d(B * 48 * 48 * 160), 256, 0, stream>>>(g1cat, P[27], P[28], P[29], g1cat, B, 48, 48, 160, 48, 144, 96, 144);
  k_avgpool2<<<g1d(B * 144 * 24 * 80), 256, 0, stream>>>(g1cat, pool1, B, 144, 48, 160);

  // --- basic b1 (144) -> cat with psv2 (192) ---
  conv({pool1, P[42], nullptr, P[43], P[44], nullptr, b1tmp,
        B, 144, 24, 80, 144, 24, 80, 3, 3, 1, 0, 144, 0, 144, 0, CONV_RELU});
  conv({b1tmp, P[45], nullptr, P[46], P[47], pool1, cat192,
        B, 144, 24, 80, 144, 24, 80, 3, 3, 1, 0, 144, 0, 192, 144, CONV_RELU});
  k_copy_chan<<<g1d(B * 48 * 24 * 80), 256, 0, stream>>>(c3b, cat192, B, 48, 24, 80, 144, 192);

  // --- ghost2 (192 -> 576) ---
  k_copy_chan<<<g1d(B * 192 * 24 * 80), 256, 0, stream>>>(cat192, g2cat, B, 192, 24, 80, 0, 576);
  conv({cat192, P[30], nullptr, P[31], P[32], nullptr, g2cat,
        B, 192, 24, 80, 192, 24, 80, 3, 3, 1, 0, 192, 192, 576, 0, CONV_RELU});
  k_dwconv<<<g1d(B * 192 * 24 * 80), 256, 0, stream>>>(g2cat, P[33], P[34], P[35], g2cat, B, 192, 24, 80, 192, 576, 384, 576);

  // --- basic b2 (576) -> psv16 written straight into features concat ---
  conv({g2cat, P[48], nullptr, P[49], P[50], nullptr, b2tmp,
        B, 576, 24, 80, 576, 24, 80, 3, 3, 1, 0, 576, 0, 576, 0, CONV_RELU});
  conv({b2tmp, P[51], nullptr, P[52], P[53], g2cat, outF,
        B, 576, 24, 80, 576, 24, 80, 3, 3, 1, 0, 576, 128, 704, 576, CONV_RELU});
  k_copy_chan<<<g1d(B * 128 * 24 * 80), 256, 0, stream>>>(left2, outF, B, 128, 24, 80, 0, 704);

  // --- depth head: fused up2 + conv (bilinear gather inside im2col) ---
  conv({outF, P[54], P[55], P[56], P[57], nullptr, d1out,
        B, 576, 24, 80, 288, 48, 160, 3, 3, 1, 128, 704, 0, 288, 0, CONV_RELU | CONV_UP2});
  conv({d1out, P[58], P[59], P[60], P[61], nullptr, d2out,
        B, 288, 48, 160, 144, 96, 320, 3, 3, 1, 0, 288, 0, 144, 0, CONV_RELU | CONV_UP2});
  conv({d2out, P[62], P[63], nullptr, nullptr, nullptr, outD,
        B, 144, 96, 320, 48, 96, 320, 1, 1, 0, 0, 144, 0, 48, 0, 0});
}
